// NeuralMem_41205916238481
// MI455X (gfx1250) — matrix operand order repaired and checked by reading, not compiled, not run on hardware
//
#include <hip/hip_runtime.h>

// ---------------- problem constants ----------------
#define IH 128
#define IW 128
#define IC 3
#define KH 32
#define KW 32
#define PADc 10
#define LHc 117
#define LWc 117
#define Lc (LHc * LWc)          // 13689 patches
#define Dc 3072                 // C*KH*KW
#define NMEM 4096
#define MT 856                  // ceil(L/16) M-tiles
#define KTILES 96               // 3072 / 32
#define NTILES 256              // 4096 / 16
#define LPAD (MT * 16)          // 13696

// ---------------- workspace layout (bytes) ----------------
static const size_t OFF_BIAS = 0;                                   // 4096 f32
static const size_t SZ_BIAS  = (size_t)NMEM * 4;
static const size_t OFF_B    = OFF_BIAS + SZ_BIAS;                  // packed B bf16
static const size_t SZ_B     = (size_t)NTILES * KTILES * 32 * 16 * 2;
static const size_t OFF_A    = OFF_B + SZ_B;                        // packed A bf16
static const size_t SZ_A     = (size_t)MT * KTILES * 32 * 16 * 2;
static const size_t OFF_IDS  = OFF_A + SZ_A;                        // LPAD i32
static const size_t SZ_IDS   = (size_t)LPAD * 4;
static const size_t OFF_FOLD = OFF_IDS + SZ_IDS;                    // 3*128*128 f32
static const size_t SZ_FOLD  = (size_t)IC * IH * IW * 4;
static const size_t OFF_PMAX = OFF_FOLD + SZ_FOLD;                  // 192 f32
static const size_t SZ_PMAX  = 192 * 4;
static const size_t OFF_MAX  = OFF_PMAX + SZ_PMAX;                  // 1 f32

// ---------------- WMMA types ----------------
typedef __bf16 bf16_t;
typedef bf16_t v16bf __attribute__((ext_vector_type(16)));
typedef float  v8f   __attribute__((ext_vector_type(8)));

__device__ __forceinline__ unsigned short f2bf(float f) {
    unsigned int u = __float_as_uint(f);
    unsigned int r = u + 0x7FFFu + ((u >> 16) & 1u);   // round-to-nearest-even
    return (unsigned short)(r >> 16);
}

// ---------------- bias[n] = -0.5 * ||mem[n]||^2 ----------------
__global__ __launch_bounds__(128) void bias_kernel(const float* __restrict__ mem,
                                                   float* __restrict__ bias) {
    __shared__ float s[128];
    const int n = blockIdx.x;
    float acc = 0.f;
    for (int d = threadIdx.x; d < Dc; d += 128) {
        float v = mem[(size_t)n * Dc + d];
        acc += v * v;
    }
    s[threadIdx.x] = acc;
    __syncthreads();
    for (int o = 64; o > 0; o >>= 1) {
        if (threadIdx.x < o) s[threadIdx.x] += s[threadIdx.x + o];
        __syncthreads();
    }
    if (threadIdx.x == 0) bias[n] = -0.5f * s[0];
}

// ---------------- pack B: mem -> bf16 WMMA fragments ----------------
// B tile (32x16, KxN): lane = N%16; elements 0..7 -> K 8h..8h+7, 8..15 -> K 16+8h..16+8h+7.
__global__ __launch_bounds__(256) void packB_kernel(const float* __restrict__ mem,
                                                    unsigned short* __restrict__ Bpk) {
    long long t = (long long)blockIdx.x * 256 + threadIdx.x;
    if (t >= (long long)NTILES * KTILES * 32) return;
    const int  lane = (int)(t & 31);
    long long  tile = t >> 5;
    const int  kt = (int)(tile % KTILES);
    const int  nt = (int)(tile / KTILES);
    const int  n  = nt * 16 + (lane & 15);
    const int  kb = kt * 32 + ((lane < 16) ? 0 : 8);
    const float* src = mem + (size_t)n * Dc + kb;
    unsigned short* dst = Bpk + ((size_t)tile * 32 + lane) * 16;
#pragma unroll
    for (int e = 0; e < 16; ++e) dst[e] = f2bf(src[e + ((e >= 8) ? 8 : 0)]);
}

// ---------------- pack A: unfold image -> bf16 WMMA fragments ----------------
// A tile (16x32, MxK), 16-bit layout: lane = M%16; per-lane element e -> K via
// documented interleave (VGPR0..3 hold K 0..7 / 8..15, VGPR4..7 hold K 16..23 / 24..31).
__global__ __launch_bounds__(256) void packA_kernel(const float* __restrict__ image,
                                                    unsigned short* __restrict__ Apk) {
    long long t = (long long)blockIdx.x * 256 + threadIdx.x;
    if (t >= (long long)MT * KTILES * 32) return;
    const int  lane = (int)(t & 31);
    long long  tile = t >> 5;
    const int  kt = (int)(tile % KTILES);
    const int  mt = (int)(tile / KTILES);
    const int  m  = mt * 16 + (lane & 15);
    unsigned short* dst = Apk + ((size_t)tile * 32 + lane) * 16;
    if (m >= Lc) {
#pragma unroll
        for (int e = 0; e < 16; ++e) dst[e] = 0;
        return;
    }
    const int lh = m / LWc, lw = m % LWc;
    const int khalf = (lane < 16) ? 0 : 8;
#pragma unroll
    for (int e = 0; e < 16; ++e) {
        const int v = e >> 1, hb = e & 1;
        int kl = (v < 4) ? (2 * v + hb) : (16 + 2 * (v - 4) + hb);
        kl += khalf;
        const int k   = kt * 32 + kl;
        const int c   = k >> 10;          // k / 1024
        const int rem = k & 1023;
        const int kh  = rem >> 5;
        const int kw  = rem & 31;
        const int r   = lh + kh - PADc;
        const int cc  = lw + kw - PADc;
        float val = 0.f;
        if (r >= 0 && r < IH && cc >= 0 && cc < IW)
            val = image[((size_t)r * IW + cc) * IC + c];
        dst[e] = f2bf(val);
    }
}

// ---------------- fused GEMM + bias + argmax + mapping gather ----------------
// 256 threads = 8 waves; wave w owns M-tile (blockIdx*8+w). B tiles for the
// current 4-N-tile group are staged in LDS with double-buffered
// GLOBAL_LOAD_ASYNC_TO_LDS_B128 (ASYNCcnt pipeline), shared by all 8 waves.
#define KCHUNK 4                                    // k-tiles per LDS stage
#define NCHUNKS (KTILES / KCHUNK)                   // 24
#define FRAGS_PER_CHUNK (KCHUNK * 4 * 32)           // 512 v16bf = 16 KB

__global__ __launch_bounds__(256) void gemm_argmax_kernel(
    const unsigned short* __restrict__ Apk_u, const unsigned short* __restrict__ Bpk_u,
    const float* __restrict__ bias, const int* __restrict__ mapping,
    int* __restrict__ ids) {
    __shared__ v16bf sB[2][FRAGS_PER_CHUNK];   // 32 KB, double-buffered B stage
    __shared__ float sBest[8][16][16];         // 8 KB
    __shared__ int   sIdx[8][16][16];          // 8 KB

    const int tid  = threadIdx.x;
    const int wave = tid >> 5;
    const int lane = tid & 31;
    const int mt   = blockIdx.x * 8 + wave;
    const int ncol = lane & 15;

    const v16bf* __restrict__ Ap =
        reinterpret_cast<const v16bf*>(Apk_u) + (size_t)mt * KTILES * 32 + lane;
    const char* __restrict__ Bbytes = reinterpret_cast<const char*>(Bpk_u);

    float best[8];
    int   bidx[8];
#pragma unroll
    for (int i = 0; i < 8; ++i) { best[i] = -3.402823e38f; bidx[i] = 0; }

    // Issue one 16 KB B chunk (4 k-tiles x 4 N-tiles) as 4 async b128 per thread.
    auto issue_chunk = [&](int g, int ch, int buf) {
#pragma unroll
        for (int it = 0; it < 4; ++it) {
            const int idx  = it * 256 + tid;       // 0..1023 (16-byte units)
            const int half = idx & 1;
            const int f    = idx >> 1;             // fragment index in chunk
            const int ktl  = f >> 7;               // k-tile within chunk
            const int j    = (f >> 5) & 3;         // N-tile within group
            const int ln   = f & 31;               // fragment lane
            const char* src = Bbytes +
                ((((size_t)(g * 4 + j) * KTILES + (ch * KCHUNK + ktl)) * 32 + ln) * 32 +
                 (size_t)half * 16);
            unsigned int       ldsa = (unsigned int)(size_t)((char*)&sB[buf][f] + half * 16);
            unsigned long long ga   = (unsigned long long)(size_t)src;
            asm volatile("global_load_async_to_lds_b128 %0, %1, off"
                         :: "v"(ldsa), "v"(ga)
                         : "memory");
        }
    };

    for (int g = 0; g < NTILES / 4; ++g) {
        v8f acc[4];
#pragma unroll
        for (int j = 0; j < 4; ++j)
            acc[j] = (v8f){0.f, 0.f, 0.f, 0.f, 0.f, 0.f, 0.f, 0.f};

        issue_chunk(g, 0, 0);
        for (int ch = 0; ch < NCHUNKS; ++ch) {
            const int buf = ch & 1;
            if (ch + 1 < NCHUNKS) {
                issue_chunk(g, ch + 1, buf ^ 1);
                asm volatile("s_wait_asynccnt 0x4" ::: "memory");  // chunk ch landed
            } else {
                asm volatile("s_wait_asynccnt 0x0" ::: "memory");
            }
            __syncthreads();   // all waves' async writes visible

#pragma unroll
            for (int ktl = 0; ktl < KCHUNK; ++ktl) {
                const int kt = ch * KCHUNK + ktl;
                v16bf a = Ap[(size_t)kt * 32];
                __builtin_prefetch(Ap + (size_t)(kt + 8) * 32, 0, 0);  // global_prefetch_b8
#pragma unroll
                for (int j = 0; j < 4; ++j) {
                    v16bf b = sB[buf][(ktl * 4 + j) * 32 + lane];       // ds_load_b128 x2
                    acc[j] = __builtin_amdgcn_wmma_f32_16x16x32_bf16(
                        false, a, false, b, (short)0, acc[j], false, false);
                }
            }
            __syncthreads();   // buffer reuse guard
        }

        // running argmax: VGPR i holds row (i + (lane>=16)*8), column lane%16
#pragma unroll
        for (int j = 0; j < 4; ++j) {
            const int n    = (g * 4 + j) * 16 + ncol;
            const float bb = bias[n];
#pragma unroll
            for (int i = 0; i < 8; ++i) {
                float s = acc[j][i] + bb;
                if (s > best[i]) { best[i] = s; bidx[i] = n; }  // strict > keeps first max
            }
        }
    }

    const int rbase = (lane < 16) ? 0 : 8;
#pragma unroll
    for (int i = 0; i < 8; ++i) {
        sBest[wave][rbase + i][ncol] = best[i];
        sIdx[wave][rbase + i][ncol]  = bidx[i];
    }
    __syncthreads();

    if (lane < 16) {
        const int row = lane;
        float bs = sBest[wave][row][0];
        int   bi = sIdx[wave][row][0];
#pragma unroll
        for (int c = 1; c < 16; ++c) {
            float s  = sBest[wave][row][c];
            int   ix = sIdx[wave][row][c];
            if (s > bs || (s == bs && ix < bi)) { bs = s; bi = ix; }
        }
        const int m = mt * 16 + row;
        if (m < Lc) ids[m] = mapping[bi];   // fused pattern-id gather
    }
}

// ---------------- fold: gather-accumulate overlaps ----------------
__global__ __launch_bounds__(256) void fold_kernel(const float* __restrict__ mem2,
                                                   const int* __restrict__ ids,
                                                   float* __restrict__ folded) {
    const int t = blockIdx.x * 256 + threadIdx.x;
    if (t >= IC * IH * IW) return;
    const int w = t % IW;
    const int h = (t / IW) % IH;
    const int c = t / (IH * IW);
    const int hp = h + PADc, wp = w + PADc;
    const int kh0 = max(0, hp - (LHc - 1)), kh1 = min(KH - 1, hp);
    const int kw0 = max(0, wp - (LWc - 1)), kw1 = min(KW - 1, wp);
    float acc = 0.f;
    for (int kh = kh0; kh <= kh1; ++kh) {
        const int lh = hp - kh;
        const int* idrow = ids + lh * LWc;
        const int dkbase = c * (KH * KW) + kh * KW;
        for (int kw = kw0; kw <= kw1; ++kw) {
            const int lw  = wp - kw;
            const int pid = idrow[lw];
            acc += mem2[(size_t)pid * Dc + dkbase + kw];
        }
    }
    folded[t] = acc;
}

// ---------------- global max, two stages ----------------
__global__ __launch_bounds__(256) void max1_kernel(const float* __restrict__ folded,
                                                   float* __restrict__ pmax) {
    __shared__ float s[256];
    const int t = blockIdx.x * 256 + threadIdx.x;   // exactly 192*256 = 49152 threads
    s[threadIdx.x] = folded[t];
    __syncthreads();
    for (int o = 128; o > 0; o >>= 1) {
        if (threadIdx.x < o) s[threadIdx.x] = fmaxf(s[threadIdx.x], s[threadIdx.x + o]);
        __syncthreads();
    }
    if (threadIdx.x == 0) pmax[blockIdx.x] = s[0];
}

__global__ __launch_bounds__(256) void max2_kernel(const float* __restrict__ pmax,
                                                   float* __restrict__ maxv) {
    __shared__ float s[256];
    s[threadIdx.x] = (threadIdx.x < 192) ? pmax[threadIdx.x] : -3.402823e38f;
    __syncthreads();
    for (int o = 128; o > 0; o >>= 1) {
        if (threadIdx.x < o) s[threadIdx.x] = fmaxf(s[threadIdx.x], s[threadIdx.x + o]);
        __syncthreads();
    }
    if (threadIdx.x == 0) maxv[0] = s[0];
}

// ---------------- normalize + CHW -> HWC ----------------
__global__ __launch_bounds__(256) void norm_kernel(const float* __restrict__ folded,
                                                   const float* __restrict__ maxv,
                                                   float* __restrict__ out) {
    const int t = blockIdx.x * 256 + threadIdx.x;
    if (t >= IC * IH * IW) return;
    const int w = t % IW;
    const int h = (t / IW) % IH;
    const int c = t / (IH * IW);
    out[((size_t)h * IW + w) * IC + c] = folded[t] / maxv[0];
}

// ---------------- launcher ----------------
extern "C" void kernel_launch(void* const* d_in, const int* in_sizes, int n_in,
                              void* d_out, int out_size, void* d_ws, size_t ws_size,
                              hipStream_t stream) {
    const float* image   = (const float*)d_in[0];   // (128,128,3)
    const float* mem     = (const float*)d_in[1];   // (4096,3072)
    const float* mem2    = (const float*)d_in[2];   // (4096,3072)
    const int*   mapping = (const int*)d_in[3];     // (4096,)
    float*       out     = (float*)d_out;           // (128,128,3)

    char* ws = (char*)d_ws;
    float*          bias = (float*)(ws + OFF_BIAS);
    unsigned short* Bpk  = (unsigned short*)(ws + OFF_B);
    unsigned short* Apk  = (unsigned short*)(ws + OFF_A);
    int*            ids  = (int*)(ws + OFF_IDS);
    float*          fold = (float*)(ws + OFF_FOLD);
    float*          pmax = (float*)(ws + OFF_PMAX);
    float*          maxv = (float*)(ws + OFF_MAX);

    // 1) column bias
    bias_kernel<<<NMEM, 128, 0, stream>>>(mem, bias);

    // 2) pack B (mem -> bf16 WMMA fragments)
    {
        long long total = (long long)NTILES * KTILES * 32;
        int blocks = (int)((total + 255) / 256);
        packB_kernel<<<blocks, 256, 0, stream>>>(mem, Bpk);
    }

    // 3) pack A (fused unfold -> bf16 WMMA fragments)
    {
        long long total = (long long)MT * KTILES * 32;
        int blocks = (int)((total + 255) / 256);
        packA_kernel<<<blocks, 256, 0, stream>>>(image, Apk);
    }

    // 4) fused GEMM (13696x4096x3072) + argmax + mapping gather
    gemm_argmax_kernel<<<MT / 8, 256, 0, stream>>>(Apk, Bpk, bias, mapping, ids);

    // 5) fold (overlap-sum reconstruct)
    fold_kernel<<<(IC * IH * IW + 255) / 256, 256, 0, stream>>>(mem2, ids, fold);

    // 6) global max
    max1_kernel<<<192, 256, 0, stream>>>(fold, pmax);
    max2_kernel<<<1, 256, 0, stream>>>(pmax, maxv);

    // 7) normalize + layout transform
    norm_kernel<<<(IC * IH * IW + 255) / 256, 256, 0, stream>>>(fold, maxv, out);
}
